// GATLayer_55009941127938
// MI455X (gfx1250) — matrix operand order repaired and checked by reading, not compiled, not run on hardware
//
#include <hip/hip_runtime.h>
#include <stdint.h>
#include <stddef.h>

// ---------- CDNA5 WMMA types ----------
typedef __attribute__((ext_vector_type(16))) __bf16 v16bf;
typedef __attribute__((ext_vector_type(16))) short  v16s;
typedef __attribute__((ext_vector_type(8)))  float  v8f;

union BF16Vec { v16s s; v16bf bf; };

#define IN_DIM  128
#define OUT_DIM 64
#define D_NBR   16

__device__ __forceinline__ short f32_to_bf16_rne(float f) {
  uint32_t u = __float_as_uint(f);
  u += 0x7FFFu + ((u >> 16) & 1u);   // round-to-nearest-even (inputs finite)
  return (short)(u >> 16);
}

// =====================================================================
// Kernel A: z[N,64] = h[N,128] @ W_fc[64,128]^T  via v_wmma_f32_16x16x32_bf16
// One wave per 16-row M tile. A fragments kept in VGPRs across the 4 N-tiles.
// Layouts used here (contiguous K; A and B share it):
//   A (16x32 bf16): lane<16 holds row M=lane, K = 0..15; lane>=16 same
//                   row, K = 16..31. Vector elem order = K ascending
//                   within each half.
//   B (32x16 bf16): lane<16 holds col N=lane, K=0..15; lane>=16 col N=lane-16,
//                   K=16..31 (16 contiguous floats of a W_fc row).
//   C/D (16x16 f32): VGPR v, lane<16 -> (M=v, N=lane); lane>=16 -> (M=v+8).
// =====================================================================
__global__ __launch_bounds__(256) void gat_gemm_z(const float* __restrict__ h,
                                                  const float* __restrict__ Wfc,
                                                  float* __restrict__ z,
                                                  int n_mtiles) {
  const int lane  = threadIdx.x & 31;
  const int wavei = threadIdx.x >> 5;
  const int mtile = blockIdx.x * 8 + wavei;
  if (mtile >= n_mtiles) return;   // wave-uniform -> EXEC all-ones inside

  const int row0 = mtile * 16;
  const int r    = lane & 15;
  const int hi   = lane >> 4;

  // ---- A fragments: 4 K-slabs of 32 ----
  v16bf afrag[4];
#pragma unroll
  for (int kk = 0; kk < 4; ++kk) {
    const float* p = h + (size_t)(row0 + r) * IN_DIM + kk * 32 + hi * 16;
    BF16Vec a;
#pragma unroll
    for (int e = 0; e < 8; ++e) a.s[e]     = f32_to_bf16_rne(p[e]);
#pragma unroll
    for (int e = 0; e < 8; ++e) a.s[8 + e] = f32_to_bf16_rne(p[8 + e]);
    afrag[kk] = a.bf;
  }

  // ---- 4 N-tiles of 16 output columns ----
#pragma unroll
  for (int t = 0; t < 4; ++t) {
    v8f acc = {};
#pragma unroll
    for (int kk = 0; kk < 4; ++kk) {
      const float* q = Wfc + (size_t)(t * 16 + r) * IN_DIM + kk * 32 + hi * 16;
      BF16Vec b;
#pragma unroll
      for (int e = 0; e < 16; ++e) b.s[e] = f32_to_bf16_rne(q[e]);
      acc = __builtin_amdgcn_wmma_f32_16x16x32_bf16(
          /*neg_a=*/false, afrag[kk], /*neg_b=*/false, b.bf,
          /*c_mod=*/(short)0, acc, /*reuse_a=*/false, /*reuse_b=*/false);
    }
    float* zo = z + (size_t)(row0 + hi * 8) * OUT_DIM + t * 16 + r;
#pragma unroll
    for (int v = 0; v < 8; ++v) zo[(size_t)v * OUT_DIM] = acc[v];
  }
}

// =====================================================================
// Kernel B: one wave32 per node. Lane l owns feature columns {2l, 2l+1}.
// Gather 16 neighbor rows (float2/lane, b64 loads, issued back-to-back),
// wave-reduce the attention dots, exact sort-based entmax-1.5 over 16
// values fully unrolled in registers, then alpha-weighted sum.
// =====================================================================
__device__ __forceinline__ float wave_sum32(float v) {
#pragma unroll
  for (int off = 16; off >= 1; off >>= 1) v += __shfl_xor(v, off, 32);
  return v;
}

__global__ __launch_bounds__(256) void gat_attn(const float* __restrict__ z,
                                                const int*   __restrict__ nbr,
                                                const float* __restrict__ w,
                                                const float* __restrict__ Wattn,
                                                float* __restrict__ out,
                                                int N) {
  const int lane = threadIdx.x & 31;
  const int i    = (int)((blockIdx.x * blockDim.x + threadIdx.x) >> 5);
  if (i >= N) return;

  const float2 asrc = *(const float2*)(Wattn + 2 * lane);
  const float2 adst = *(const float2*)(Wattn + 64 + 2 * lane);
  const float2 zi   = *(const float2*)(z + (size_t)i * OUT_DIM + 2 * lane);
  const float  edst = wave_sum32(zi.x * adst.x + zi.y * adst.y);

  // --- gather neighbor indices then all 16 rows (latency hiding) ---
  int jidx[D_NBR];
#pragma unroll
  for (int d = 0; d < D_NBR; ++d) jidx[d] = nbr[(size_t)i * D_NBR + d];

  float2 zs[D_NBR];
#pragma unroll
  for (int d = 0; d < D_NBR; ++d)
    zs[d] = *(const float2*)(z + (size_t)jidx[d] * OUT_DIM + 2 * lane);

  // --- attention logits: e = leaky_relu(z_src.a_src + z.a_dst) + w ---
  float x[D_NBR];
  float xmax = -3.0e38f;
#pragma unroll
  for (int d = 0; d < D_NBR; ++d) {
    float p = wave_sum32(zs[d].x * asrc.x + zs[d].y * asrc.y) + edst;
    p = (p >= 0.0f) ? p : 0.01f * p;            // leaky_relu, slope 0.01
    p += w[(size_t)i * D_NBR + d];
    p *= 0.5f;                                   // entmax: x = e/2
    x[d] = p;
    xmax = fmaxf(xmax, p);
  }
#pragma unroll
  for (int d = 0; d < D_NBR; ++d) x[d] -= xmax;

  // --- sort descending: odd-even transposition network, fully unrolled ---
  float xs[D_NBR];
#pragma unroll
  for (int d = 0; d < D_NBR; ++d) xs[d] = x[d];
#pragma unroll
  for (int pass = 0; pass < D_NBR; ++pass) {
#pragma unroll
    for (int k = (pass & 1); k + 1 < D_NBR; k += 2) {
      float a = xs[k], b = xs[k + 1];
      xs[k] = fmaxf(a, b);
      xs[k + 1] = fminf(a, b);
    }
  }

  // --- tau(k), support size, tau* ---
  float tau[D_NBR];
  float cum = 0.0f, cumsq = 0.0f;
#pragma unroll
  for (int k = 0; k < D_NBR; ++k) {
    cum   += xs[k];
    cumsq += xs[k] * xs[k];
    const float kk    = (float)(k + 1);
    const float rk    = 1.0f / kk;
    const float mean  = cum * rk;
    const float msq   = cumsq * rk;
    const float ss    = kk * (msq - mean * mean);
    const float delta = (1.0f - ss) * rk;
    const float sq    = (delta > 0.0f) ? __builtin_sqrtf(delta) : 0.0f;
    tau[k] = mean - sq;
  }
  int support = 0;
#pragma unroll
  for (int k = 0; k < D_NBR; ++k) support += (tau[k] <= xs[k]) ? 1 : 0;
  float tau_star = tau[0];
#pragma unroll
  for (int k = 1; k < D_NBR; ++k)
    tau_star = (support == k + 1) ? tau[k] : tau_star;

  // --- output: sum_d alpha_d * z_src_d ---
  float2 acc; acc.x = 0.0f; acc.y = 0.0f;
#pragma unroll
  for (int d = 0; d < D_NBR; ++d) {
    float y = fmaxf(x[d] - tau_star, 0.0f);
    float a = y * y;
    acc.x += a * zs[d].x;
    acc.y += a * zs[d].y;
  }
  *(float2*)(out + (size_t)i * OUT_DIM + 2 * lane) = acc;
}

// =====================================================================
extern "C" void kernel_launch(void* const* d_in, const int* in_sizes, int n_in,
                              void* d_out, int out_size, void* d_ws, size_t ws_size,
                              hipStream_t stream) {
  const float* h     = (const float*)d_in[0];  // (N,128)
  const int*   nbr   = (const int*)  d_in[1];  // (N,16)
  const float* w     = (const float*)d_in[2];  // (N,16,1)
  const float* Wfc   = (const float*)d_in[3];  // (64,128)
  const float* Wattn = (const float*)d_in[4];  // (1,128)
  float*       out   = (float*)d_out;          // (N,64)
  float*       z     = (float*)d_ws;           // scratch: (N,64) f32

  const int N        = in_sizes[0] / IN_DIM;   // 100000
  const int n_mtiles = (N + 15) / 16;          // 6250 (exact)

  gat_gemm_z<<<(n_mtiles + 7) / 8, 256, 0, stream>>>(h, Wfc, z, n_mtiles);

  const long long threads = (long long)N * 32;
  gat_attn<<<(unsigned)((threads + 255) / 256), 256, 0, stream>>>(
      z, nbr, w, Wattn, out, N);
}
